// MultiHeadAttention_89232240542292
// MI455X (gfx1250) — matrix operand order repaired and checked by reading, not compiled, not run on hardware
//
#include <hip/hip_runtime.h>
#include <hip/hip_bf16.h>

#define BATCH 2
#define SEQ 2048
#define DMODEL 1024
#define NHEAD 16
#define HDIM 64

typedef __bf16 bf16;
typedef __bf16 v16bf  __attribute__((ext_vector_type(16)));
typedef __bf16 bf16x8 __attribute__((ext_vector_type(8)));
typedef __bf16 bf16x4 __attribute__((ext_vector_type(4)));
typedef float  v8f    __attribute__((ext_vector_type(8)));
typedef unsigned int u32x4 __attribute__((ext_vector_type(4)));
typedef int i32x4 __attribute__((ext_vector_type(4)));
typedef int i32x8 __attribute__((ext_vector_type(8)));

__device__ __forceinline__ v16bf make_frag(bf16x8 lo, bf16x8 hi) {
  v16bf r;
#pragma unroll
  for (int i = 0; i < 8; ++i) { r[i] = lo[i]; r[i + 8] = hi[i]; }
  return r;
}

__device__ __forceinline__ v8f wmma_bf16(v16bf a, v16bf b, v8f c) {
  // D = A(16x32 bf16) * B(32x16 bf16) + C(16x16 f32)
  return __builtin_amdgcn_wmma_f32_16x16x32_bf16(
      /*neg_a=*/false, a, /*neg_b=*/false, b,
      /*c_mod=*/(short)0, c, /*reuse_a=*/false, /*reuse_b=*/false);
}

// CDNA5 async DMA: global -> LDS, 16 bytes per lane, tracked by ASYNCcnt.
__device__ __forceinline__ void async_copy_b128(void* lds, const void* g) {
  const unsigned lds_off = (unsigned)(uintptr_t)lds;
  asm volatile("global_load_async_to_lds_b128 %0, %1, off"
               :
               : "v"(lds_off), "v"(g)
               : "memory");
}

__device__ __forceinline__ void wait_async_zero() {
  asm volatile("s_wait_asynccnt 0x0" ::: "memory");
}

// CDNA5 Tensor Data Mover: DMA a 2D tile (data_size = 2 bytes) into LDS with
// hardware row padding.  D# packing per CDNA5 ISA.
// This toolchain's builtin is the 6-arg form:
//   (u32x4 g0, i32x8 g1, i32x4 g2, i32x4 g3, i32x8, i32 cpol)
__device__ __forceinline__ void tdm_load_2d_bf16(
    const void* lds, const void* gaddr,
    unsigned tensor_d0, unsigned tensor_d1,   // tensor extents (elements)
    unsigned tile_d0, unsigned tile_d1,       // tile extents  (elements)
    unsigned d0_stride,                       // dim0 stride   (elements)
    unsigned pad_interval, unsigned pad_amount) {
  const unsigned long long ga = (unsigned long long)(uintptr_t)gaddr;
  u32x4 g0;
  g0[0] = 1u;                                     // count=1, user descriptor
  g0[1] = (unsigned)(uintptr_t)lds;               // lds_addr (bytes)
  g0[2] = (unsigned)(ga & 0xFFFFFFFFu);           // global_addr[31:0]
  g0[3] = (unsigned)((ga >> 32) & 0x01FFFFFFu)    // global_addr[56:32]
          | (2u << 30);                           // type = 2 ("image")
  i32x8 g1;
  g1[0] = (int)((1u << 16)                        // data_size = 2 bytes
                | (1u << 20)                      // pad_enable
                | (pad_interval << 22) | (pad_amount << 25));
  g1[1] = (int)((tensor_d0 & 0xFFFFu) << 16);                       // dim0 lo16
  g1[2] = (int)(((tensor_d0 >> 16) & 0xFFFFu) |
                ((tensor_d1 & 0xFFFFu) << 16));                     // dim0 hi | dim1 lo
  g1[3] = (int)(((tensor_d1 >> 16) & 0xFFFFu) |
                ((tile_d0 & 0xFFFFu) << 16));                       // dim1 hi | tile0
  g1[4] = (int)(tile_d1 & 0xFFFFu);                                 // tile1 (tile2=0)
  g1[5] = (int)d0_stride;                                           // dim0 stride lo32
  g1[6] = 0;                                                        // stride hi / dim1 stride
  g1[7] = 0;
  const i32x4 z4 = {0, 0, 0, 0};                  // groups 2/3 unused (2D)
  const i32x8 z8 = {0, 0, 0, 0, 0, 0, 0, 0};
  __builtin_amdgcn_tensor_load_to_lds(g0, g1, z4, z4, z8, 0);
}

__device__ __forceinline__ bf16x4 cvt4(float4 f) {
  bf16x4 r;
  r[0] = (bf16)f.x; r[1] = (bf16)f.y; r[2] = (bf16)f.z; r[3] = (bf16)f.w;
  return r;
}

// ---------------------------------------------------------------------------
// Kernel 1: fused QKV projection.  out = x @ W^T + bias, cast to bf16.
// blockIdx.z selects {Q,K,V}.  Q,K stored [B,H,S,HD]; V stored [B,H,HD,S].
// ---------------------------------------------------------------------------
__global__ __launch_bounds__(256)
void qkv_proj_kernel(const float* __restrict__ x,
                     const float* __restrict__ Wq, const float* __restrict__ bq,
                     const float* __restrict__ Wk, const float* __restrict__ bk,
                     const float* __restrict__ Wv, const float* __restrict__ bv,
                     bf16* __restrict__ Qd, bf16* __restrict__ Kd,
                     bf16* __restrict__ Vtd) {
  const int which = blockIdx.z;
  const float* W    = (which == 0) ? Wq : (which == 1) ? Wk : Wv;
  const float* bias = (which == 0) ? bq : (which == 1) ? bk : bv;

  __shared__ bf16 xs[128][40];  // 128 token-rows x 32 k
  __shared__ bf16 ws[64][40];   // 64 out-cols  x 32 k

  const int tid  = threadIdx.x;
  const int lane = tid & 31;
  const int wid  = tid >> 5;   // 0..7
  const int wr   = wid >> 1;   // 0..3 : 32-row strip
  const int wc   = wid & 1;    // 0..1 : 32-col strip

  const int row0 = blockIdx.x * 128;
  const int col0 = blockIdx.y * 64;

  const int m  = lane & 15;
  const int ka = (lane < 16) ? 0 : 8;
  const int kb = (lane < 16) ? 0 : 8;
  const int m0 = (lane < 16) ? 0 : 8;

  const int xi = tid >> 3, xj = (tid & 7) << 2;

  v8f acc[2][2];
#pragma unroll
  for (int i = 0; i < 2; ++i)
#pragma unroll
    for (int j = 0; j < 2; ++j) acc[i][j] = {};

  for (int k0 = 0; k0 < DMODEL; k0 += 32) {
    __syncthreads();
    float4 fx[4];
#pragma unroll
    for (int e = 0; e < 4; ++e)
      fx[e] = *(const float4*)&x[(size_t)(row0 + xi + e * 32) * DMODEL + k0 + xj];
    float4 fw[2];
#pragma unroll
    for (int e = 0; e < 2; ++e)
      fw[e] = *(const float4*)&W[(size_t)(col0 + xi + e * 32) * DMODEL + k0 + xj];
#pragma unroll
    for (int e = 0; e < 4; ++e)
      *(bf16x4*)&xs[xi + e * 32][xj] = cvt4(fx[e]);
#pragma unroll
    for (int e = 0; e < 2; ++e)
      *(bf16x4*)&ws[xi + e * 32][xj] = cvt4(fw[e]);
    __syncthreads();

    v16bf a[2], b[2];
#pragma unroll
    for (int rt = 0; rt < 2; ++rt) {
      const bf16* p = &xs[wr * 32 + rt * 16 + m][0];
      a[rt] = make_frag(*(const bf16x8*)&p[ka], *(const bf16x8*)&p[ka + 16]);
    }
#pragma unroll
    for (int ct = 0; ct < 2; ++ct) {
      const bf16* p = &ws[wc * 32 + ct * 16 + m][0];
      b[ct] = make_frag(*(const bf16x8*)&p[kb], *(const bf16x8*)&p[kb + 16]);
    }
#pragma unroll
    for (int rt = 0; rt < 2; ++rt)
#pragma unroll
      for (int ct = 0; ct < 2; ++ct)
        acc[rt][ct] = wmma_bf16(a[rt], b[ct], acc[rt][ct]);
  }

#pragma unroll
  for (int rt = 0; rt < 2; ++rt) {
#pragma unroll
    for (int ct = 0; ct < 2; ++ct) {
      const int dcol = col0 + wc * 32 + ct * 16 + m;
      const int h    = dcol >> 6;
      const int hd   = dcol & 63;
      const float bb = bias[dcol];
      const int grow0 = row0 + wr * 32 + rt * 16 + m0;
      const int bidx  = grow0 / SEQ;
      const int s0    = grow0 - bidx * SEQ;
      if (which < 2) {
        bf16* dst = (which == 0) ? Qd : Kd;
        bf16* p = &dst[((size_t)(bidx * NHEAD + h) * SEQ + s0) * HDIM + hd];
#pragma unroll
        for (int i = 0; i < 8; ++i)
          p[(size_t)i * HDIM] = (bf16)(acc[rt][ct][i] + bb);
      } else {
        bf16x8 pk;
#pragma unroll
        for (int i = 0; i < 8; ++i) pk[i] = (bf16)(acc[rt][ct][i] + bb);
        *(bf16x8*)&Vtd[((size_t)(bidx * NHEAD + h) * HDIM + hd) * SEQ + s0] = pk;
      }
    }
  }
}

// ---------------------------------------------------------------------------
// Kernel 2: flash attention with TDM double buffering.
// Wave 0 drives tensor_load_to_lds (TENSORcnt); in-order completion lets us
// prefetch block i+1 while computing block i (wait tensorcnt<=2).
// ---------------------------------------------------------------------------
__global__ __launch_bounds__(128)
void flash_attn_kernel(const bf16* __restrict__ Qd, const bf16* __restrict__ Kd,
                       const bf16* __restrict__ Vtd, bf16* __restrict__ ctx) {
  __shared__ bf16 Ks[2][32][72];   // double-buffered: 32 keys x 64 hd (+pad)
  __shared__ bf16 Vs[2][64][40];   // double-buffered: 64 hd x 32 keys (+pad)
  __shared__ bf16 Ps[4][16][40];   // per-wave probs: 16 q x 32 keys

  const int tid  = threadIdx.x;
  const int lane = tid & 31;
  const int wid  = tid >> 5;      // 0..3
  const int bh   = blockIdx.y;
  const int bb   = bh / NHEAD;
  const int h    = bh % NHEAD;
  const int q0   = blockIdx.x * 64 + wid * 16;

  const bf16* Qp  = Qd  + (size_t)bh * SEQ * HDIM;
  const bf16* Kp  = Kd  + (size_t)bh * SEQ * HDIM;
  const bf16* Vtp = Vtd + (size_t)bh * HDIM * SEQ;

  const int m  = lane & 15;
  const int ka = (lane < 16) ? 0 : 8;
  const int kb = (lane < 16) ? 0 : 8;
  const int m0 = (lane < 16) ? 0 : 8;

  // Q fragments for this wave's 16 queries
  const bf16* qrow = &Qp[(size_t)(q0 + m) * HDIM];
  const v16bf aq0 = make_frag(*(const bf16x8*)&qrow[ka],
                              *(const bf16x8*)&qrow[ka + 16]);
  const v16bf aq1 = make_frag(*(const bf16x8*)&qrow[32 + ka],
                              *(const bf16x8*)&qrow[32 + ka + 16]);

  const float kscale = 0.18033688011112042f;  // (1/sqrt(64)) * log2(e)

  v8f o[4];
#pragma unroll
  for (int t = 0; t < 4; ++t) o[t] = {};
  float mrun[8], l[8];
#pragma unroll
  for (int i = 0; i < 8; ++i) { mrun[i] = -1e30f; l[i] = 0.0f; }

  // prologue: kick off TDM for key-block 0 into buffer 0
  if (wid == 0) {
    // K tile: rows=keys (dim1), cols=hd (dim0); row = 32 dwords -> pad 4 dwords
    tdm_load_2d_bf16(&Ks[0][0][0], Kp, HDIM, SEQ, HDIM, 32, HDIM, 4u, 3u);
    // V^T tile: rows=hd (dim1), cols=keys (dim0); row = 16 dwords -> pad 4
    tdm_load_2d_bf16(&Vs[0][0][0], Vtp, SEQ, HDIM, 32, HDIM, SEQ, 3u, 3u);
  }

  for (int kb0 = 0; kb0 < SEQ; kb0 += 32) {
    const int buf = (kb0 >> 5) & 1;
    if (wid == 0) {
      if (kb0 + 32 < SEQ) {  // prefetch next block, then wait for current
        const int nb = buf ^ 1;
        tdm_load_2d_bf16(&Ks[nb][0][0], Kp + (size_t)(kb0 + 32) * HDIM,
                         HDIM, SEQ, HDIM, 32, HDIM, 4u, 3u);
        tdm_load_2d_bf16(&Vs[nb][0][0], Vtp + (kb0 + 32),
                         SEQ, HDIM, 32, HDIM, SEQ, 3u, 3u);
        __builtin_amdgcn_s_wait_tensorcnt(2);  // current block complete
      } else {
        __builtin_amdgcn_s_wait_tensorcnt(0);
      }
    }
    __syncthreads();  // current block visible to all waves

    // scores S = Q @ K^T for 32 keys
    v8f s[2];
#pragma unroll
    for (int nt = 0; nt < 2; ++nt) {
      const bf16* kr = &Ks[buf][nt * 16 + m][0];
      const v16bf b0 = make_frag(*(const bf16x8*)&kr[kb],
                                 *(const bf16x8*)&kr[kb + 16]);
      const v16bf b1 = make_frag(*(const bf16x8*)&kr[32 + kb],
                                 *(const bf16x8*)&kr[32 + kb + 16]);
      v8f z = {};
      z = wmma_bf16(aq0, b0, z);
      s[nt] = wmma_bf16(aq1, b1, z);
    }

    // online softmax
    float p0[8], p1[8];
#pragma unroll
    for (int i = 0; i < 8; ++i) {
      float mb = fmaxf(s[0][i], s[1][i]);
#pragma unroll
      for (int off = 1; off < 16; off <<= 1)
        mb = fmaxf(mb, __shfl_xor(mb, off, 32));
      const float mn   = fmaxf(mrun[i], mb);
      const float corr = exp2f((mrun[i] - mn) * kscale);
      const float a0 = exp2f((s[0][i] - mn) * kscale);
      const float a1 = exp2f((s[1][i] - mn) * kscale);
      float rs = a0 + a1;
#pragma unroll
      for (int off = 1; off < 16; off <<= 1)
        rs += __shfl_xor(rs, off, 32);
      l[i]   = l[i] * corr + rs;
      mrun[i] = mn;
      p0[i] = a0; p1[i] = a1;
#pragma unroll
      for (int t = 0; t < 4; ++t) o[t][i] *= corr;
    }

    // reshape P (C-layout) -> A-fragment layout through LDS
#pragma unroll
    for (int i = 0; i < 8; ++i) {
      Ps[wid][m0 + i][m]      = (bf16)p0[i];
      Ps[wid][m0 + i][m + 16] = (bf16)p1[i];
    }
    asm volatile("s_wait_dscnt 0x0" ::: "memory");  // wave-local LDS RAW fence

    // ctx += P(16x32) @ V(32x64)
    const bf16* pr = &Ps[wid][m][0];
    const v16bf aP = make_frag(*(const bf16x8*)&pr[ka],
                               *(const bf16x8*)&pr[ka + 16]);
#pragma unroll
    for (int t = 0; t < 4; ++t) {
      const bf16* vr = &Vs[buf][t * 16 + m][0];
      const v16bf bV = make_frag(*(const bf16x8*)&vr[kb],
                                 *(const bf16x8*)&vr[kb + 16]);
      o[t] = wmma_bf16(aP, bV, o[t]);
    }
    __syncthreads();  // reads done: buffer may become next TDM target
  }

  // normalize and store ctx as bf16 in [B,S,D]
  float inv[8];
#pragma unroll
  for (int i = 0; i < 8; ++i) inv[i] = 1.0f / l[i];
  const int srow = q0 + m0;
#pragma unroll
  for (int t = 0; t < 4; ++t) {
    const int dcol = h * HDIM + t * 16 + m;
    bf16* p = &ctx[((size_t)bb * SEQ + srow) * DMODEL + dcol];
#pragma unroll
    for (int i = 0; i < 8; ++i)
      p[(size_t)i * DMODEL] = (bf16)(o[t][i] * inv[i]);
  }
}

// ---------------------------------------------------------------------------
// Kernel 3: output projection.  out = ctx @ Wo^T + bo  (f32 output)
// ctx staged via per-lane async global->LDS DMA (ASYNCcnt path).
// ---------------------------------------------------------------------------
__global__ __launch_bounds__(256)
void out_proj_kernel(const bf16* __restrict__ ctx, const float* __restrict__ Wo,
                     const float* __restrict__ bo, float* __restrict__ out) {
  __shared__ bf16 cs[128][40];
  __shared__ bf16 ws[64][40];

  const int tid  = threadIdx.x;
  const int lane = tid & 31;
  const int wid  = tid >> 5;
  const int wr   = wid >> 1;
  const int wc   = wid & 1;

  const int row0 = blockIdx.x * 128;
  const int col0 = blockIdx.y * 64;

  const int m  = lane & 15;
  const int ka = (lane < 16) ? 0 : 8;
  const int kb = (lane < 16) ? 0 : 8;
  const int m0 = (lane < 16) ? 0 : 8;

  const int ci = tid >> 2, cj = (tid & 3) << 3;   // ctx: 16B chunks
  const int wi = tid >> 3, wj = (tid & 7) << 2;   // W: float4

  v8f acc[2][2];
#pragma unroll
  for (int i = 0; i < 2; ++i)
#pragma unroll
    for (int j = 0; j < 2; ++j) acc[i][j] = {};

  for (int k0 = 0; k0 < DMODEL; k0 += 32) {
    __syncthreads();
#pragma unroll
    for (int e = 0; e < 2; ++e)
      async_copy_b128(&cs[ci + e * 64][cj],
                      &ctx[(size_t)(row0 + ci + e * 64) * DMODEL + k0 + cj]);
    float4 fw[2];
#pragma unroll
    for (int e = 0; e < 2; ++e)
      fw[e] = *(const float4*)&Wo[(size_t)(col0 + wi + e * 32) * DMODEL + k0 + wj];
#pragma unroll
    for (int e = 0; e < 2; ++e)
      *(bf16x4*)&ws[wi + e * 32][wj] = cvt4(fw[e]);
    wait_async_zero();
    __syncthreads();

    v16bf a[2], b[2];
#pragma unroll
    for (int rt = 0; rt < 2; ++rt) {
      const bf16* p = &cs[wr * 32 + rt * 16 + m][0];
      a[rt] = make_frag(*(const bf16x8*)&p[ka], *(const bf16x8*)&p[ka + 16]);
    }
#pragma unroll
    for (int ct = 0; ct < 2; ++ct) {
      const bf16* p = &ws[wc * 32 + ct * 16 + m][0];
      b[ct] = make_frag(*(const bf16x8*)&p[kb], *(const bf16x8*)&p[kb + 16]);
    }
#pragma unroll
    for (int rt = 0; rt < 2; ++rt)
#pragma unroll
      for (int ct = 0; ct < 2; ++ct)
        acc[rt][ct] = wmma_bf16(a[rt], b[ct], acc[rt][ct]);
  }

#pragma unroll
  for (int rt = 0; rt < 2; ++rt) {
#pragma unroll
    for (int ct = 0; ct < 2; ++ct) {
      const int dcol  = col0 + wc * 32 + ct * 16 + m;
      const float bbv = bo[dcol];
      const int grow0 = row0 + wr * 32 + rt * 16 + m0;
      float* p = &out[(size_t)grow0 * DMODEL + dcol];
#pragma unroll
      for (int i = 0; i < 8; ++i)
        p[(size_t)i * DMODEL] = acc[rt][ct][i] + bbv;
    }
  }
}

// ---------------------------------------------------------------------------
extern "C" void kernel_launch(void* const* d_in, const int* in_sizes, int n_in,
                              void* d_out, int out_size, void* d_ws,
                              size_t ws_size, hipStream_t stream) {
  const float* x  = (const float*)d_in[0];
  const float* Wq = (const float*)d_in[1];
  const float* bq = (const float*)d_in[2];
  const float* Wk = (const float*)d_in[3];
  const float* bk = (const float*)d_in[4];
  const float* Wv = (const float*)d_in[5];
  const float* bv = (const float*)d_in[6];
  const float* Wo = (const float*)d_in[7];
  const float* bo = (const float*)d_in[8];
  float* out = (float*)d_out;

  const size_t qkv_elems = (size_t)BATCH * NHEAD * SEQ * HDIM;  // 4M
  bf16* Qd   = (bf16*)d_ws;
  bf16* Kd   = Qd + qkv_elems;
  bf16* Vtd  = Kd + qkv_elems;
  bf16* ctxb = Vtd + qkv_elems;  // total ws = 32 MB

  dim3 g1((BATCH * SEQ) / 128, DMODEL / 64, 3);
  qkv_proj_kernel<<<g1, 256, 0, stream>>>(x, Wq, bq, Wk, bk, Wv, bv,
                                          Qd, Kd, Vtd);

  dim3 g2(SEQ / 64, BATCH * NHEAD, 1);
  flash_attn_kernel<<<g2, 128, 0, stream>>>(Qd, Kd, Vtd, ctxb);

  dim3 g3((BATCH * SEQ) / 128, DMODEL / 64, 1);
  out_proj_kernel<<<g3, 256, 0, stream>>>(ctxb, Wo, bo, out);
}
